// ScaledDotProductAttention_50259707298262
// MI455X (gfx1250) — matrix operand order repaired and checked by reading, not compiled, not run on hardware
//
#include <hip/hip_runtime.h>
#include <hip/hip_bf16.h>

typedef __attribute__((ext_vector_type(16))) _Float16 v16h;
typedef __attribute__((ext_vector_type(8)))  float    v8f;

#define NH    16
#define SQ    2048
#define SK    2048
#define DH    64
#define QTILE 16

#define S_ELEMS (QTILE * SK)            // 32768 floats: score strip
#define O_ELEMS (QTILE * DH)            // 1024 floats: output tile
#define SMEM_FLOATS (S_ELEMS + O_ELEMS + 256 + 32)
#define SMEM_BYTES  (SMEM_FLOATS * sizeof(float))

// One workgroup = 256 threads = 8 waves; handles a 16-query tile of one head.
__global__ __launch_bounds__(256)
void sdpa_wmma_kernel(const float* __restrict__ qg,
                      const float* __restrict__ kg,
                      const float* __restrict__ vg,
                      const float* __restrict__ mg,
                      float* __restrict__ out_g,
                      float* __restrict__ attn_g)
{
    extern __shared__ float smem[];
    float* sS    = smem;                 // [16][2048] masked/scaled scores -> P
    float* sO    = sS + S_ELEMS;         // [16][64]  accumulated output
    float* sPart = sO + O_ELEMS;         // [256]     reduction partials
    float* sRow  = sPart + 256;          // [0..15]=rowmax, [16..31]=1/rowsum

    const int t    = threadIdx.x;
    const int wave = t >> 5;
    const int lane = t & 31;
    const int nn   = lane & 15;          // N index / A row index
    const int hi   = lane >> 4;          // half-wave selector
    const int h    = blockIdx.y;
    const int q0   = blockIdx.x * QTILE;

    // zero the output accumulator tile (used much later; barriers below cover it)
    for (int i = t; i < O_ELEMS; i += 256) sO[i] = 0.0f;

    // ---------------- load Q tile into WMMA A-layout registers ----------------
    // 16-bit A 16x32: halves i<8 -> K = hi*8 + i ; halves i>=8 -> K = 16 + hi*8 + (i-8)
    const float* qrow = qg + ((size_t)(h * SQ + q0 + nn) * DH);
    v16h A0, A1;
    #pragma unroll
    for (int i = 0; i < 8; ++i) {
        A0[i]     = (_Float16)qrow[ 0 + hi * 8 + i];
        A0[8 + i] = (_Float16)qrow[16 + hi * 8 + i];
        A1[i]     = (_Float16)qrow[32 + hi * 8 + i];
        A1[8 + i] = (_Float16)qrow[48 + hi * 8 + i];
    }

    // ---------------- S = (Q K^T) * scale, masked -> LDS ----------------------
    const float scale = 0.125f;  // 1/sqrt(64)
    const float* kbase = kg + (size_t)h * SK * DH;
    const float* mbase = mg + ((size_t)(h * SQ + q0) * (size_t)SK);

    for (int kt = 0; kt < 16; ++kt) {
        const int key0 = wave * 256 + kt * 16;
        // B 32x16 f16: lane holds column nn; half i<8 -> K = hi*8 + i ; half i>=8 -> K = 16 + hi*8 + (i-8)
        const float* krow = kbase + (size_t)(key0 + nn) * DH;
        v16h B0, B1;
        #pragma unroll
        for (int i = 0; i < 16; ++i) {
            B0[i] = (_Float16)krow[ 0 + (i >> 3) * 16 + hi * 8 + (i & 7)];
            B1[i] = (_Float16)krow[32 + (i >> 3) * 16 + hi * 8 + (i & 7)];
        }
        v8f c = {};
        c = __builtin_amdgcn_wmma_f32_16x16x32_f16(false, A0, false, B0,
                                                   (short)0, c, false, false);
        c = __builtin_amdgcn_wmma_f32_16x16x32_f16(false, A1, false, B1,
                                                   (short)0, c, false, false);
        // C/D layout: VGPR j -> M = j + hi*8, N = nn
        #pragma unroll
        for (int j = 0; j < 8; ++j) {
            const int m = j + hi * 8;
            const float mv = mbase[(size_t)m * SK + key0 + nn];
            const float s  = (mv == 0.0f) ? -1000.0f : c[j] * scale;
            sS[m * SK + key0 + nn] = s;
        }
    }
    __syncthreads();

    // ---------------- softmax over SK in LDS ----------------------------------
    const int r  = t >> 4;   // 16 threads per row
    const int cc = t & 15;
    float mval = -3.402823466e38f;
    for (int cI = cc; cI < SK; cI += 16) mval = fmaxf(mval, sS[r * SK + cI]);
    sPart[t] = mval;
    __syncthreads();
    if (cc == 0) {
        float mm = sPart[t];
        #pragma unroll
        for (int i = 1; i < 16; ++i) mm = fmaxf(mm, sPart[t + i]);
        sRow[r] = mm;
    }
    __syncthreads();
    const float rmax = sRow[r];
    float ssum = 0.0f;
    for (int cI = cc; cI < SK; cI += 16) {
        const float e = __expf(sS[r * SK + cI] - rmax);
        sS[r * SK + cI] = e;
        ssum += e;
    }
    sPart[t] = ssum;
    __syncthreads();
    if (cc == 0) {
        float s2 = sPart[t];
        #pragma unroll
        for (int i = 1; i < 16; ++i) s2 += sPart[t + i];
        sRow[16 + r] = 1.0f / s2;
    }
    __syncthreads();

    // normalize in LDS and stream attn to HBM fully coalesced (dominant write)
    float* attn_base = attn_g + ((size_t)(h * SQ + q0) * SK);
    for (int idx = t; idx < S_ELEMS; idx += 256) {
        const int rr = idx >> 11;                 // idx / 2048
        const float val = sS[idx] * sRow[16 + rr];
        sS[idx] = val;
        attn_base[idx] = val;
    }
    __syncthreads();

    // ---------------- O = P @ V ------------------------------------------------
    const float* vbase = vg + (size_t)h * SK * DH;
    v8f acc[4] = {};
    for (int ck = 0; ck < 8; ++ck) {
        const int kk = wave * 256 + ck * 32;
        // A from LDS: row nn, keys kk + hi*16 + i (contiguous 16 per half-wave)
        v16h Ap;
        const float* ap = &sS[nn * SK + kk + hi * 16];
        #pragma unroll
        for (int i = 0; i < 8; ++i) {
            Ap[i]     = (_Float16)ap[i];
            Ap[8 + i] = (_Float16)ap[8 + i];
        }
        #pragma unroll
        for (int nt = 0; nt < 4; ++nt) {
            // B from V: (k, n) = V[kk + i + hi*16][nt*16 + nn]
            v16h Bv;
            const float* vp = vbase + (size_t)(kk + hi * 16) * DH + nt * 16 + nn;
            #pragma unroll
            for (int i = 0; i < 16; ++i)
                Bv[i] = (_Float16)vp[(size_t)i * DH];
            acc[nt] = __builtin_amdgcn_wmma_f32_16x16x32_f16(false, Ap, false, Bv,
                                                             (short)0, acc[nt],
                                                             false, false);
        }
    }
    // cross-wave reduction of partial O via LDS float atomics (ds_add_f32)
    #pragma unroll
    for (int nt = 0; nt < 4; ++nt) {
        #pragma unroll
        for (int j = 0; j < 8; ++j)
            atomicAdd(&sO[(j + hi * 8) * DH + nt * 16 + nn], acc[nt][j]);
    }
    __syncthreads();

    float* obase = out_g + ((size_t)(h * SQ + q0) * DH);
    for (int idx = t; idx < O_ELEMS; idx += 256)
        obase[idx] = sO[idx];
}

extern "C" void kernel_launch(void* const* d_in, const int* in_sizes, int n_in,
                              void* d_out, int out_size, void* d_ws, size_t ws_size,
                              hipStream_t stream) {
    (void)in_sizes; (void)n_in; (void)out_size; (void)d_ws; (void)ws_size;
    const float* q = (const float*)d_in[0];
    const float* k = (const float*)d_in[1];
    const float* v = (const float*)d_in[2];
    const float* m = (const float*)d_in[3];
    float* out_g  = (float*)d_out;                       // (1,16,2048,64)
    float* attn_g = out_g + (size_t)NH * SQ * DH;        // (1,16,2048,2048)

    hipFuncSetAttribute((const void*)sdpa_wmma_kernel,
                        hipFuncAttributeMaxDynamicSharedMemorySize,
                        (int)SMEM_BYTES);

    dim3 grid(SQ / QTILE, NH);
    dim3 block(256);
    sdpa_wmma_kernel<<<grid, block, SMEM_BYTES, stream>>>(q, k, v, m, out_g, attn_g);
}
